// DamageSegmentor_54906861912326
// MI455X (gfx1250) — matrix operand order repaired and checked by reading, not compiled, not run on hardware
//
#include <hip/hip_runtime.h>
#include <hip/hip_bf16.h>

#define HW      262144      // 512*512
#define NPIX    2097152     // 8*512*512
#define NEGV    (-1000000000.0f)
#define INV_TEMP 10.0f

typedef __attribute__((ext_vector_type(16))) __bf16 v16bf_t;
typedef __attribute__((ext_vector_type(8)))  __bf16 v8bf_t;
typedef __attribute__((ext_vector_type(8)))  float  v8f_t;

// acc layout: [0]=ce_sum [1]=focal_sum [2]=grad_sum [3]=per_sum [4..8]=inter [9..13]=denom [14]=cls_loss
// ints layout: [0..7]=fg_per_img [8]=nt_total [9]=n_t [10]=n_o

__device__ __forceinline__ float wave_sum32(float v) {
#pragma unroll
    for (int d = 1; d < 32; d <<= 1) v += __shfl_xor(v, d, 32);
    return v;
}

// ---------------- Kernel 1: fused softmax / CE / focal / dice / fg-prob map ----------------
__global__ __launch_bounds__(256) void ds_pixel_kernel(const float* __restrict__ logits,
                                                       const int* __restrict__ tgt,
                                                       float* __restrict__ P,
                                                       float* __restrict__ acc,
                                                       int* __restrict__ ints) {
    __shared__ float s_acc[12];
    int t = threadIdx.x;
    if (t < 12) s_acc[t] = 0.0f;
    __syncthreads();

    int i = blockIdx.x * 256 + t;
    int b = i >> 18, hw = i & (HW - 1);
    const float* base = logits + (size_t)b * 5 * HW + hw;
    float x[5];
#pragma unroll
    for (int c = 0; c < 5; ++c) x[c] = base[(size_t)c * HW];
    float m = fmaxf(fmaxf(fmaxf(x[0], x[1]), fmaxf(x[2], x[3])), x[4]);
    float e[5], se = 0.0f;
#pragma unroll
    for (int c = 0; c < 5; ++c) { e[c] = __expf(x[c] - m); se += e[c]; }
    float inv = 1.0f / se;
    float lse = __logf(se);
    int tc = tgt[i];
    float ce = -(x[tc] - m - lse);
    float pt = __expf(-ce);
    float om = fminf(fmaxf(1.0f - pt, 0.0f), 1.0f);
    float focal = om * om * ce;             // GAMMA = 2
    float pc[5];
#pragma unroll
    for (int c = 0; c < 5; ++c) pc[c] = e[c] * inv;
    P[i] = pc[1];                           // FG_IDX = 1
    if (tc == 1) atomicAdd(&ints[b], 1);

    float vals[12];
    vals[0] = ce; vals[1] = focal;
#pragma unroll
    for (int c = 0; c < 5; ++c) {
        vals[2 + c] = (c == tc) ? pc[c] : 0.0f;                 // inter
        vals[7 + c] = pc[c] + ((c == tc) ? 1.0f : 0.0f);        // denom
    }
#pragma unroll
    for (int j = 0; j < 12; ++j) {
        float v = wave_sum32(vals[j]);
        if ((t & 31) == 0) atomicAdd(&s_acc[j], v);
    }
    __syncthreads();
    if (t < 12) {
        int g = (t < 2) ? t : (t + 2);      // 0,1 -> ce,focal ; 2..6 -> inter(4..8) ; 7..11 -> denom(9..13)
        atomicAdd(&acc[g], s_acc[t]);
    }
}

// ---------------- Kernel 2: |sobel(P) - sobel(target_fg)| mean ----------------
__global__ __launch_bounds__(256) void ds_sobel_kernel(const float* __restrict__ P,
                                                       const int* __restrict__ tgt,
                                                       float* __restrict__ acc) {
    const float kx[3][3] = {{-1,0,1},{-2,0,2},{-1,0,1}};
    const float ky[3][3] = {{-1,-2,-1},{0,0,0},{1,2,1}};
    int i = blockIdx.x * 256 + threadIdx.x;
    int b = i >> 18, hw = i & (HW - 1);
    int y = hw >> 9, x = hw & 511;
    float gpx = 0, gpy = 0, gtx = 0, gty = 0;
#pragma unroll
    for (int dy = -1; dy <= 1; ++dy)
#pragma unroll
        for (int dx = -1; dx <= 1; ++dx) {
            int yy = y + dy, xx = x + dx;
            float p = 0.0f, tf = 0.0f;
            if (yy >= 0 && yy < 512 && xx >= 0 && xx < 512) {
                int idx = (b << 18) + (yy << 9) + xx;
                p = P[idx];
                tf = (tgt[idx] > 0) ? 1.0f : 0.0f;
            }
            float wx = kx[dy + 1][dx + 1], wy = ky[dy + 1][dx + 1];
            gpx += wx * p; gpy += wy * p; gtx += wx * tf; gty += wy * tf;
        }
    float gp = sqrtf(gpx * gpx + gpy * gpy + 1e-6f);
    float gt = sqrtf(gtx * gtx + gty * gty + 1e-6f);
    float v = wave_sum32(fabsf(gp - gt));
    __shared__ float sb;
    if (threadIdx.x == 0) sb = 0.0f;
    __syncthreads();
    if ((threadIdx.x & 31) == 0) atomicAdd(&sb, v);
    __syncthreads();
    if (threadIdx.x == 0) atomicAdd(&acc[2], sb);
}

// ---------------- Kernels 3-5: ordered first-N index selection ----------------
__device__ __forceinline__ bool is_tiny(const int* tgt, const int* fgc, int i) {
    int b = i >> 18;
    int fg = fgc[b];
    return (tgt[i] == 1) && (fg > 0) && (fg <= 2048);
}

__global__ __launch_bounds__(256) void ds_count_kernel(const int* __restrict__ tgt,
                                                       const int* __restrict__ ints,
                                                       int* __restrict__ blk_cnt) {
    __shared__ int s[256];
    int t = threadIdx.x;
    int base = blockIdx.x * 1024 + t * 4;
    int c = 0;
#pragma unroll
    for (int k = 0; k < 4; ++k) c += is_tiny(tgt, ints, base + k) ? 1 : 0;
    s[t] = c; __syncthreads();
    for (int off = 128; off > 0; off >>= 1) {
        if (t < off) s[t] += s[t + off];
        __syncthreads();
    }
    if (t == 0) blk_cnt[blockIdx.x] = s[0];
}

__global__ __launch_bounds__(256) void ds_prefix_kernel(const int* __restrict__ blk_cnt,
                                                        int* __restrict__ blk_off,
                                                        int* __restrict__ ints) {
    __shared__ int s[256];
    int t = threadIdx.x;
    int local[8], sum = 0;
#pragma unroll
    for (int j = 0; j < 8; ++j) { local[j] = blk_cnt[t * 8 + j]; sum += local[j]; }
    s[t] = sum; __syncthreads();
    for (int off = 1; off < 256; off <<= 1) {
        int v = (t >= off) ? s[t - off] : 0;
        __syncthreads();
        s[t] += v;
        __syncthreads();
    }
    int run = s[t] - sum;   // exclusive
    for (int j = 0; j < 8; ++j) { blk_off[t * 8 + j] = run; run += local[j]; }
    if (t == 255) {
        int tot = s[255];
        ints[8] = tot;
        int nt = tot < 1024 ? tot : 1024;
        int no = (NPIX - tot) < 4096 ? (NPIX - tot) : 4096;
        ints[9] = nt; ints[10] = no;
    }
}

__global__ __launch_bounds__(256) void ds_select_kernel(const int* __restrict__ tgt,
                                                        const int* __restrict__ ints,
                                                        const int* __restrict__ blk_off,
                                                        int* __restrict__ tiny_idx,
                                                        int* __restrict__ other_idx) {
    __shared__ int s[256];
    int t = threadIdx.x;
    int base = blockIdx.x * 1024 + t * 4;
    bool f[4]; int c = 0;
#pragma unroll
    for (int k = 0; k < 4; ++k) { f[k] = is_tiny(tgt, ints, base + k); c += f[k] ? 1 : 0; }
    s[t] = c; __syncthreads();
    for (int off = 1; off < 256; off <<= 1) {
        int v = (t >= off) ? s[t - off] : 0;
        __syncthreads();
        s[t] += v;
        __syncthreads();
    }
    int trank = blk_off[blockIdx.x] + (s[t] - c);   // tiny elements strictly before my first item
#pragma unroll
    for (int k = 0; k < 4; ++k) {
        int i = base + k;
        if (f[k]) {
            if (trank < 1024) tiny_idx[trank] = i;
            trank++;
        } else {
            int orank = i - trank;                  // non-tiny elements before i
            if (orank < 4096) other_idx[orank] = i;
        }
    }
}

// ---------------- Kernel 6: gather + L2-normalize -> bf16 embeddings ----------------
__global__ __launch_bounds__(128) void ds_gather_kernel(const float* __restrict__ pf,
                                                        const int* __restrict__ ints,
                                                        const int* __restrict__ tiny_idx,
                                                        const int* __restrict__ other_idx,
                                                        __bf16* __restrict__ Tn,
                                                        __bf16* __restrict__ On) {
    int w = threadIdx.x >> 5, ln = threadIdx.x & 31;
    int slot = blockIdx.x * 4 + w;                  // 0..5119
    int n_t = ints[9], n_o = ints[10];
    int idx; __bf16* dst;
    if (slot < 1024) {
        if (slot >= n_t) return;
        idx = tiny_idx[slot]; dst = Tn + slot * 32;
    } else {
        int ss = slot - 1024;
        if (ss >= n_o) return;
        idx = other_idx[ss]; dst = On + ss * 32;
    }
    int b = idx >> 18, hw = idx & (HW - 1);
    float v = pf[((size_t)b * 32 + ln) * HW + hw];
    float ssq = wave_sum32(v * v);
    float inv = 1.0f / fmaxf(sqrtf(ssq), 1e-12f);
    dst[ln] = (__bf16)(v * inv);
}

// ---------------- Kernel 7: WMMA Gram tiles + fused row-logsumexp ----------------
// One wave per 16-row tile of t_emb. Two phases: 64 sim_t tiles -> st[], then
// 256 sim_o tiles -> so[], each unrolled x8 (16 static v_wmma_f32_16x16x32_bf16,
// no per-tile source/accumulator selects). Fixed-shift LSE: sim/TEMP <= 10 so
// exp(sim/TEMP - 10) never overflows; LSE = 10 + log(sum). The sim_t diagonal
// is removed once after the loop instead of per-tile masking.
__global__ __launch_bounds__(32) void ds_wmma_lse_kernel(const __bf16* __restrict__ Tn,
                                                         const __bf16* __restrict__ On,
                                                         const int* __restrict__ ints,
                                                         float* __restrict__ acc) {
    const int l = threadIdx.x;
    const int half = l >> 4, mh = l & 15;
    const int RA = blockIdx.x * 16;
    const int n_t = ints[9], n_o = ints[10];

    // A fragment: per-lane = one contiguous 32B chunk, k = 16*half + s (matches B)
    const __bf16* arow = Tn + (RA + mh) * 32;
    v16bf_t a = *(const v16bf_t*)(arow + 16 * half);

    float st[8], so[8];
#pragma unroll
    for (int r = 0; r < 8; ++r) { st[r] = 0.0f; so[r] = 0.0f; }

    // ---- Phase 1: sim_t (t_emb @ t_emb^T), 64 column tiles ----
    for (int ct0 = 0; ct0 < 64; ct0 += 8) {
        __builtin_prefetch(Tn + ((ct0 + 8) * 16 + mh) * 32, 0, 0);  // speculative; OOB dropped
#pragma unroll
        for (int u = 0; u < 8; ++u) {
            const int cb = (ct0 + u) * 16;
            const float cs = ((cb + mh) < n_t) ? 1.0f : 0.0f;
            v16bf_t bf = *(const v16bf_t*)(Tn + (cb + mh) * 32 + 16 * half);
            v8f_t cm = {};
            cm = __builtin_amdgcn_wmma_f32_16x16x32_bf16(false, a, false, bf,
                                                         (short)0, cm, false, false);
#pragma unroll
            for (int r = 0; r < 8; ++r)
                st[r] += __expf(cm[r] * INV_TEMP - 10.0f) * cs;
        }
    }

    // ---- Phase 2: sim_o (t_emb @ o_emb^T), 256 column tiles ----
    for (int ct0 = 0; ct0 < 256; ct0 += 8) {
        __builtin_prefetch(On + ((ct0 + 8) * 16 + mh) * 32, 0, 0);
#pragma unroll
        for (int u = 0; u < 8; ++u) {
            const int cb = (ct0 + u) * 16;
            const float cs = ((cb + mh) < n_o) ? 1.0f : 0.0f;
            v16bf_t bf = *(const v16bf_t*)(On + (cb + mh) * 32 + 16 * half);
            v8f_t cm = {};
            cm = __builtin_amdgcn_wmma_f32_16x16x32_bf16(false, a, false, bf,
                                                         (short)0, cm, false, false);
#pragma unroll
            for (int r = 0; r < 8; ++r)
                so[r] += __expf(cm[r] * INV_TEMP - 10.0f) * cs;
        }
    }

    // Remove the sim_t diagonal term: it was accumulated exactly once, on lane
    // mh == r + 8*half of tile cb == RA. Recompute p_ii from Tn directly.
    if (((mh >> 3) == half) && (RA + mh) < n_t) {
        const __bf16* drow = Tn + (RA + mh) * 32;
        float dot = 0.0f;
#pragma unroll
        for (int k = 0; k < 32; ++k) { float tv = (float)drow[k]; dot += tv * tv; }
        float pii = __expf(dot * INV_TEMP - 10.0f);
        const int dr = mh & 7;
#pragma unroll
        for (int r = 0; r < 8; ++r)
            if (r == dr) st[r] -= pii;
    }

    // Reduce the 16 column-lanes of each half; per = log_den - log_num
    float local = 0.0f;
#pragma unroll
    for (int r = 0; r < 8; ++r) {
        float num = st[r], den = st[r] + so[r];
#pragma unroll
        for (int d = 1; d < 16; d <<= 1) {
            num += __shfl_xor(num, d, 32);
            den += __shfl_xor(den, d, 32);
        }
        int row = RA + r + 8 * half;
        if (mh == 0 && row < n_t) local += __logf(den) - __logf(num);
    }
    if (mh == 0) atomicAdd(&acc[3], local);
}

// ---------------- Kernel 8: 8x8 class contrastive (+ BCE fallback) ----------------
__global__ __launch_bounds__(256) void ds_cls_kernel(const float* __restrict__ emb,
                                                     const int* __restrict__ labels,
                                                     float* __restrict__ acc) {
    __shared__ float E[8 * 256];
    __shared__ float sim[64];
    int t = threadIdx.x, w = t >> 5, ln = t & 31;
    float v[8], ss = 0.0f;
#pragma unroll
    for (int j = 0; j < 8; ++j) { v[j] = emb[w * 256 + ln + 32 * j]; ss += v[j] * v[j]; }
    ss = wave_sum32(ss);
    float inv = 1.0f / fmaxf(sqrtf(ss), 1e-12f);
#pragma unroll
    for (int j = 0; j < 8; ++j) E[w * 256 + ln + 32 * j] = v[j] * inv;
    __syncthreads();
    if (t < 64) {
        int i = t >> 3, j = t & 7;
        float d = 0.0f;
        for (int k = 0; k < 256; ++k) d += E[i * 256 + k] * E[j * 256 + k];
        sim[t] = d;
    }
    __syncthreads();
    if (t == 0) {
        int lab[8];
        for (int i = 0; i < 8; ++i) lab[i] = labels[i];
        float persum = 0.0f; int nvalid = 0; bool anyv = false;
        for (int i = 0; i < 8; ++i) {
            bool haspos = false;
            float mp = -1e30f, ma = -1e30f;
            for (int j = 0; j < 8; ++j) if (j != i) {
                float s = sim[i * 8 + j] * INV_TEMP;
                ma = fmaxf(ma, s);
                if (lab[i] == lab[j]) { haspos = true; mp = fmaxf(mp, s); }
            }
            if (haspos) {
                float sp = 0.0f, sa = 0.0f;
                for (int j = 0; j < 8; ++j) if (j != i) {
                    float s = sim[i * 8 + j] * INV_TEMP;
                    sa += __expf(s - ma);
                    if (lab[i] == lab[j]) sp += __expf(s - mp);
                }
                persum += -((mp + __logf(sp)) - (ma + __logf(sa)));
                nvalid++; anyv = true;
            }
        }
        float contrastive = persum / (float)(nvalid > 0 ? nvalid : 1);
        float bsum = 0.0f;
        for (int i = 0; i < 8; ++i)
            for (int j = 0; j < 8; ++j) if (j != i) {
                float cs = fminf(fmaxf(sim[i * 8 + j], -1.0f), 1.0f);
                float p01 = (cs + 1.0f) * 0.5f;
                float ts = (lab[i] == lab[j]) ? 1.0f : 0.0f;
                float lp = fmaxf(__logf(p01), -100.0f);
                float lq = fmaxf(__logf(1.0f - p01), -100.0f);
                bsum += -(ts * lp + (1.0f - ts) * lq);
            }
        acc[14] = anyv ? contrastive : (bsum / 56.0f);
    }
}

// ---------------- Kernel 9: finalize ----------------
__global__ void ds_finalize_kernel(const float* __restrict__ acc,
                                   const int* __restrict__ ints,
                                   float* __restrict__ out) {
    if (threadIdx.x == 0 && blockIdx.x == 0) {
        const float Np = (float)NPIX;
        float ce = acc[0] / Np;
        float focal = acc[1] / Np;
        float grad = acc[2] / Np;
        float dsum = 0.0f;
        for (int c = 0; c < 5; ++c)
            dsum += (2.0f * acc[4 + c] + 1e-6f) / (acc[9 + c] + 1e-6f);
        float dice = 1.0f - dsum / 5.0f;
        int ntt = ints[8];
        int n_t = ints[9] > 0 ? ints[9] : 1;
        float tiny = (ntt >= 16) ? (acc[3] / (float)n_t) : 0.0f;
        out[0] = ce + dice + focal + grad + tiny + acc[14];
    }
}

// ---------------- launcher ----------------
extern "C" void kernel_launch(void* const* d_in, const int* in_sizes, int n_in,
                              void* d_out, int out_size, void* d_ws, size_t ws_size,
                              hipStream_t stream) {
    (void)in_sizes; (void)n_in; (void)out_size; (void)ws_size;
    const float* logits = (const float*)d_in[0];
    const int*   tgt    = (const int*)d_in[1];
    const float* pf     = (const float*)d_in[2];
    const float* emb    = (const float*)d_in[3];
    const int*   clab   = (const int*)d_in[4];

    char* ws = (char*)d_ws;
    float* P        = (float*)ws;                         // 8,388,608 B
    char*  zb       = ws + 8388608;
    float* acc      = (float*)(zb);                       //    64 B
    int*   ints     = (int*)(zb + 64);                    //    64 B
    int*   blk_cnt  = (int*)(zb + 128);                   //  8192 B
    int*   blk_off  = (int*)(zb + 128 + 8192);            //  8192 B
    int*   tiny_idx = (int*)(zb + 128 + 16384);           //  4096 B
    int*   other_idx= (int*)(zb + 128 + 20480);           // 16384 B
    __bf16* Tn      = (__bf16*)(zb + 128 + 36864);        // 65536 B (32B aligned)
    __bf16* On      = (__bf16*)(zb + 128 + 36864 + 65536);// 262144 B (32B aligned)
    const size_t zbytes = 128 + 36864 + 65536 + 262144;   // 364,672 B

    hipMemsetAsync(zb, 0, zbytes, stream);
    ds_pixel_kernel<<<NPIX / 256, 256, 0, stream>>>(logits, tgt, P, acc, ints);
    ds_sobel_kernel<<<NPIX / 256, 256, 0, stream>>>(P, tgt, acc);
    ds_count_kernel<<<NPIX / 1024, 256, 0, stream>>>(tgt, ints, blk_cnt);
    ds_prefix_kernel<<<1, 256, 0, stream>>>(blk_cnt, blk_off, ints);
    ds_select_kernel<<<NPIX / 1024, 256, 0, stream>>>(tgt, ints, blk_off, tiny_idx, other_idx);
    ds_gather_kernel<<<5120 / 4, 128, 0, stream>>>(pf, ints, tiny_idx, other_idx, Tn, On);
    ds_wmma_lse_kernel<<<64, 32, 0, stream>>>(Tn, On, ints, acc);
    ds_cls_kernel<<<1, 256, 0, stream>>>(emb, clab, acc);
    ds_finalize_kernel<<<1, 1, 0, stream>>>(acc, ints, (float*)d_out);
}
